// RumorDetector_87617332838922
// MI455X (gfx1250) — matrix operand order repaired and checked by reading, not compiled, not run on hardware
//
#include <hip/hip_runtime.h>
#include <math.h>

// ---------------- problem constants ----------------
constexpr int NN  = 64000;   // nodes
constexpr int NPG = 500;     // nodes per graph
constexpr int GG  = 128;     // graphs
constexpr int EE  = 131072;  // edges
constexpr int DIN = 768;     // input dim
constexpr int HH  = 512;     // hidden
constexpr int CC  = 4;       // classes
constexpr int KK  = 250;     // kept nodes per graph
constexpr int KT  = DIN / 32; // 24 k-tiles of 32
constexpr float EPS = 1e-5f;

// ---------------- WMMA / async types ----------------
typedef __attribute__((ext_vector_type(16))) __bf16        v16bf;
typedef __attribute__((ext_vector_type(8)))  float         v8f;
typedef __attribute__((ext_vector_type(8)))  unsigned int  v8u;
typedef __attribute__((ext_vector_type(4)))  int           v4i;

#if __has_builtin(__builtin_amdgcn_global_load_async_to_lds_b128)
#define ASYNC_LDS 1
#else
#define ASYNC_LDS 0
#endif

__device__ __forceinline__ unsigned int f2bf_rne(float f) {
  unsigned int u = __float_as_uint(f);
  unsigned int r = u + 0x7fffu + ((u >> 16) & 1u);
  return r >> 16;
}

// ---------------- K0: x fp32 -> bf16 (vector4) ----------------
__global__ __launch_bounds__(256) void k_cvt_x(const float* __restrict__ x,
                                               unsigned short* __restrict__ xb) {
  long long i4 = (long long)blockIdx.x * 256 + threadIdx.x; // NN*DIN/4 items
  long long base = i4 * 4;
  float4 v = *(const float4*)(x + base);
  unsigned int lo = f2bf_rne(v.x) | (f2bf_rne(v.y) << 16);
  unsigned int hi = f2bf_rne(v.z) | (f2bf_rne(v.w) << 16);
  *(uint2*)(xb + base) = make_uint2(lo, hi);
}

// ---------------- K0b: pack gcn_w, column-major per k-tile ----------------
// wp[((kt*2 + half)*HH + col)*8 + v] = pack(bf16 W[k0,col], bf16 W[k0+1,col])
// with k0 = kt*32 + (v>=4?16:0) + half*8 + 2*(v&3)  (B 32x16 16-bit layout matching A:
// lanes 0-15 K=0..7,16..23; lanes 16-31 K=8..15,24..31).  Per (kt,half), a 64-col block slice is 2KB contiguous
// and lands in LDS in the same [col][v] order -> async b128 copies + ds_load_b128.
__global__ __launch_bounds__(256) void k_pack_w(const float* __restrict__ w,
                                                unsigned int* __restrict__ wp) {
  int idx = blockIdx.x * 256 + threadIdx.x;          // KT*2*HH*8 total
  int v = idx & 7;
  int col = (idx >> 3) & (HH - 1);
  int half = (idx >> 12) & 1;
  int kt = idx >> 13;
  int k0 = kt * 32 + ((v >> 2) << 4) + half * 8 + 2 * (v & 3);
  unsigned int lo = f2bf_rne(w[(size_t)k0 * HH + col]);
  unsigned int hi = f2bf_rne(w[(size_t)(k0 + 1) * HH + col]);
  wp[idx] = lo | (hi << 16);
}

// ---------------- degree kernels ----------------
__global__ __launch_bounds__(256) void k_deg_init(float* __restrict__ deg) {
  int i = blockIdx.x * 256 + threadIdx.x;
  if (i < NN) deg[i] = 1.0f;                 // self-loop contributes 1
}
__global__ __launch_bounds__(256) void k_deg_acc(const int* __restrict__ ei,
                                                 float* __restrict__ deg) {
  int e = blockIdx.x * 256 + threadIdx.x;
  if (e < EE) atomicAdd(&deg[ei[EE + e]], 1.0f);
}
__global__ __launch_bounds__(256) void k_deg_rsqrt(float* __restrict__ deg) {
  int i = blockIdx.x * 256 + threadIdx.x;
  if (i < NN) deg[i] = rsqrtf(deg[i]);       // now holds dinv
}
__global__ __launch_bounds__(256) void k_edge_coef(const int* __restrict__ ei,
                                                   const float* __restrict__ dinv,
                                                   float* __restrict__ coef) {
  int e = blockIdx.x * 256 + threadIdx.x;
  if (e < EE) coef[e] = dinv[ei[e]] * dinv[ei[EE + e]];
}

// ---------------- K2: GEMM  hw[N,H] = x @ W  (bf16 WMMA, fp32 acc) ----------------
// Block tile 128x64, 8 waves of 32x32 (2x2 WMMA).  Weight tile staged in LDS by
// async global->LDS b128 copies, double-buffered over the K loop so the copy of
// tile k+1 overlaps the 4 WMMAs of tile k.  One s_wait_asynccnt + barrier per
// k-step.
__global__ __launch_bounds__(256) void k_gemm_xw(const unsigned short* __restrict__ xb,
                                                 const unsigned int* __restrict__ wp,
                                                 float* __restrict__ hw) {
  // [buf][half][col(64)][v(8) + pad->12]: 12-dword pitch (48B, 16B aligned) so
  // 16 lanes of a half-wave hit 16 distinct banks on ds_load_b128.
  __shared__ unsigned int bs[2][2][64][12];

  const int tid  = threadIdx.x;
  const int lane = tid & 31;
  const int wid  = tid >> 5;                 // 0..7
  const int half = lane >> 4;                // 0/1
  const int l16  = lane & 15;
  const int waveM = wid >> 1;                // 0..3
  const int waveN = wid & 1;                 // 0..1
  const int rowBase  = blockIdx.x * 128 + waveM * 32;
  const int blockCol = blockIdx.y * 64;
  const int colBase  = blockCol + waveN * 32;

  // staging role of this thread: one b128 (4 dwords = half a column) per k-step
  const int sHalf = tid >> 7;                // 0/1
  const int sCol  = (tid & 127) >> 1;        // 0..63
  const int sSeg  = (tid & 1) * 4;           // 0 or 4

  auto issue = [&](int buf, int kt) {
    const unsigned int* gsrc =
        wp + ((size_t)(kt * 2 + sHalf) * HH + blockCol + sCol) * 8 + sSeg;
    unsigned int* ldst = &bs[buf][sHalf][sCol][sSeg];
#if ASYNC_LDS
    __builtin_amdgcn_global_load_async_to_lds_b128(
        (__attribute__((address_space(1))) v4i*)gsrc,
        (__attribute__((address_space(3))) v4i*)ldst, 0, 0);
#else
    uint4 t = *(const uint4*)gsrc;
    *(uint4*)ldst = t;
#endif
  };
  auto wait_async = [&]() {
#if ASYNC_LDS
#if __has_builtin(__builtin_amdgcn_s_wait_asynccnt)
    __builtin_amdgcn_s_wait_asynccnt(0);
#else
    asm volatile("s_wait_asynccnt 0" ::: "memory");
#endif
#endif
  };

  v8f acc[2][2] = {};                        // [mi][ni], 16x16 fp32 tiles
  issue(0, 0);

  for (int kt = 0; kt < KT; ++kt) {
    const int cur = kt & 1;
    wait_async();          // this wave's async copies into `cur` have landed
    __syncthreads();       // all waves: copies visible; prev buffer fully read
    if (kt + 1 < KT) issue(cur ^ 1, kt + 1); // overlap next copy with compute

    // A fragments: 16-bit A 16x32 layout (kOf = (v>=4?16:0) + 8*half + 2*(v&3));
    // contiguous per half -> compiler emits 2x global_load_b128 per fragment.
    v8u au[2];
    #pragma unroll
    for (int mi = 0; mi < 2; ++mi) {
      const unsigned short* arow =
          xb + (size_t)(rowBase + mi * 16 + l16) * DIN + kt * 32;
      __builtin_prefetch(arow + 32, 0, 1);   // next k-tile -> global_prefetch_b8
      #pragma unroll
      for (int v = 0; v < 8; ++v) {
        const int ko = ((v >> 2) << 4) + (half << 3) + ((v & 3) << 1);
        au[mi][v] = *(const unsigned int*)(arow + ko);
      }
    }

    // B fragments from LDS: 8 contiguous dwords per lane -> 2x ds_load_b128
    v8u bu[2];
    #pragma unroll
    for (int ni = 0; ni < 2; ++ni) {
      const unsigned int* bp = &bs[cur][half][waveN * 32 + ni * 16 + l16][0];
      uint4 lo = *(const uint4*)bp;
      uint4 hi = *(const uint4*)(bp + 4);
      bu[ni] = (v8u){lo.x, lo.y, lo.z, lo.w, hi.x, hi.y, hi.z, hi.w};
    }

    #pragma unroll
    for (int mi = 0; mi < 2; ++mi)
      #pragma unroll
      for (int ni = 0; ni < 2; ++ni)
        acc[mi][ni] = __builtin_amdgcn_wmma_f32_16x16x32_bf16(
            false, __builtin_bit_cast(v16bf, au[mi]),
            false, __builtin_bit_cast(v16bf, bu[ni]),
            (short)0, acc[mi][ni], false, false);
  }

  // C/D layout: lane n = l16, VGPR r -> row r + 8*half
  #pragma unroll
  for (int mi = 0; mi < 2; ++mi)
    #pragma unroll
    for (int ni = 0; ni < 2; ++ni) {
      const int col = colBase + ni * 16 + l16;
      #pragma unroll
      for (int r = 0; r < 8; ++r) {
        const int row = rowBase + mi * 16 + (half << 3) + r;
        hw[(size_t)row * HH + col] = acc[mi][ni][r];
      }
    }
}

// ---------------- K3a: agg = hw * dinv^2 + bias (self-loop term) ----------------
__global__ __launch_bounds__(256) void k_selfbias(const float* __restrict__ hw,
                                                  const float* __restrict__ dinv,
                                                  const float* __restrict__ b,
                                                  float* __restrict__ agg) {
  long long i4 = (long long)blockIdx.x * 256 + threadIdx.x; // NN*HH/4
  long long base = i4 * 4;
  int node = (int)(base / HH);
  int h = (int)(base & (HH - 1));
  float di = dinv[node]; float sc = di * di;
  float4 v = *(const float4*)(hw + base);
  float4 o;
  o.x = v.x * sc + b[h];     o.y = v.y * sc + b[h + 1];
  o.z = v.z * sc + b[h + 2]; o.w = v.w * sc + b[h + 3];
  *(float4*)(agg + base) = o;
}

// ---------------- K3b: edge scatter  agg[dst] += coef * hw[src] ----------------
__global__ __launch_bounds__(256) void k_edge_scatter(const int* __restrict__ ei,
                                                      const float* __restrict__ coef,
                                                      const float* __restrict__ hw,
                                                      float* __restrict__ agg) {
  long long gid = (long long)blockIdx.x * 256 + threadIdx.x; // EE*128
  int e = (int)(gid >> 7);
  int c = (int)(gid & 127) * 4;
  int src = ei[e], dst = ei[EE + e];
  float cf = coef[e];
  float4 v = *(const float4*)(hw + (size_t)src * HH + c);
  float* d = agg + (size_t)dst * HH + c;
  atomicAdd(d + 0, v.x * cf); atomicAdd(d + 1, v.y * cf);
  atomicAdd(d + 2, v.z * cf); atomicAdd(d + 3, v.w * cf);
}

// ---------------- K4: in-place ReLU + score dot products ----------------
__global__ __launch_bounds__(256) void k_relu_score(float* __restrict__ h,
                                                    const float* __restrict__ w1,
                                                    const float* __restrict__ w2,
                                                    const float* __restrict__ sagb,
                                                    float* __restrict__ sc,
                                                    float* __restrict__ t2) {
  int i = blockIdx.x, tx = threadIdx.x;
  float* row = h + (size_t)i * HH;
  float r1 = fmaxf(row[tx], 0.0f), r2 = fmaxf(row[tx + 256], 0.0f);
  row[tx] = r1; row[tx + 256] = r2;
  float a1 = r1 * w1[tx] + r2 * w1[tx + 256];
  float a2 = r1 * w2[tx] + r2 * w2[tx + 256];
  __shared__ float s1[256], s2[256];
  s1[tx] = a1; s2[tx] = a2; __syncthreads();
  for (int s = 128; s > 0; s >>= 1) {
    if (tx < s) { s1[tx] += s1[tx + s]; s2[tx] += s2[tx + s]; }
    __syncthreads();
  }
  if (tx == 0) { sc[i] = s1[0] + sagb[0]; t2[i] = s2[0]; }
}

// ---------------- K5: score scatter  sc[dst] += t2[src] ----------------
__global__ __launch_bounds__(256) void k_score_scatter(const int* __restrict__ ei,
                                                       const float* __restrict__ t2,
                                                       float* __restrict__ sc) {
  int e = blockIdx.x * 256 + threadIdx.x;
  if (e < EE) atomicAdd(&sc[ei[EE + e]], t2[ei[e]]);
}

// ---------------- K6: per-graph top-K via bitonic sort in LDS ----------------
__global__ __launch_bounds__(256) void k_topk(const float* __restrict__ sc,
                                              int* __restrict__ gidx,
                                              float* __restrict__ tanhv) {
  int g = blockIdx.x;
  __shared__ float sv[512];
  __shared__ int   si[512];
  for (int t = threadIdx.x; t < 512; t += 256) {
    sv[t] = (t < NPG) ? sc[g * NPG + t] : -INFINITY;
    si[t] = t;
  }
  __syncthreads();
  for (int k = 2; k <= 512; k <<= 1) {
    for (int j = k >> 1; j > 0; j >>= 1) {
      for (int t = threadIdx.x; t < 512; t += 256) {
        int ixj = t ^ j;
        if (ixj > t) {
          float va = sv[t], vb = sv[ixj];
          int ia = si[t], ib = si[ixj];
          bool bGreater = (vb > va) || (vb == va && ib < ia);
          bool dirDesc = ((t & k) == 0);
          if (dirDesc ? bGreater : !bGreater) {
            sv[t] = vb; sv[ixj] = va; si[t] = ib; si[ixj] = ia;
          }
        }
      }
      __syncthreads();
    }
  }
  for (int r = threadIdx.x; r < KK; r += 256) {
    gidx[g * KK + r]  = g * NPG + si[r];
    tanhv[g * KK + r] = tanhf(sv[r]);
  }
}

// ---------------- K7: xp = h[gidx] * tanh(vals) ----------------
__global__ __launch_bounds__(256) void k_build_xp(const float* __restrict__ h,
                                                  const int* __restrict__ gidx,
                                                  const float* __restrict__ tanhv,
                                                  float* __restrict__ xp) {
  long long gid = (long long)blockIdx.x * 256 + threadIdx.x; // GG*KK*HH/4
  int row = (int)(gid >> 7);
  int c = (int)(gid & 127) * 4;
  int node = gidx[row];
  float tv = tanhv[row];
  float4 v = *(const float4*)(h + (size_t)node * HH + c);
  v.x *= tv; v.y *= tv; v.z *= tv; v.w *= tv;
  *(float4*)(xp + (size_t)row * HH + c) = v;
}

// ---------------- K8a: BN partial column sums (32 chunks) ----------------
__global__ __launch_bounds__(256) void k_bn_stage1(const float* __restrict__ xp,
                                                   float* __restrict__ psum,
                                                   float* __restrict__ psq) {
  int chunk = blockIdx.x;                    // 0..31, 1000 rows each
  int col = blockIdx.y * 256 + threadIdx.x;  // 0..511
  float s = 0.f, q = 0.f;
  int r0 = chunk * 1000;
  for (int r = r0; r < r0 + 1000; ++r) {
    float v = xp[(size_t)r * HH + col];
    s += v; q += v * v;
  }
  psum[chunk * HH + col] = s;
  psq[chunk * HH + col]  = q;
}

// ---------------- K8b: finalize BN affine: a = rstd*gamma, b2 = beta - mu*a ----------------
__global__ __launch_bounds__(256) void k_bn_stage2(const float* __restrict__ psum,
                                                   const float* __restrict__ psq,
                                                   const float* __restrict__ gamma,
                                                   const float* __restrict__ beta,
                                                   float* __restrict__ aff) {
  int col = blockIdx.x * 256 + threadIdx.x;
  float s = 0.f, q = 0.f;
  for (int c = 0; c < 32; ++c) { s += psum[c * HH + col]; q += psq[c * HH + col]; }
  const float invM = 1.0f / (float)(GG * KK);
  float mu = s * invM;
  float var = q * invM - mu * mu;
  float a = rsqrtf(var + EPS) * gamma[col];
  aff[col] = a;
  aff[HH + col] = beta[col] - mu * a;
}

// ---------------- K9: pooled[g,h] = rowmean*a + b2 ----------------
__global__ __launch_bounds__(256) void k_pool(const float* __restrict__ xp,
                                              const float* __restrict__ aff,
                                              float* __restrict__ pooled) {
  int g = blockIdx.x;
  int col = blockIdx.y * 256 + threadIdx.x;
  float s = 0.f;
  const float* base = xp + (size_t)g * KK * HH + col;
  for (int r = 0; r < KK; ++r) s += base[(size_t)r * HH];
  pooled[(size_t)g * HH + col] = (s / (float)KK) * aff[col] + aff[HH + col];
}

// ---------------- K10: FC + log_softmax ----------------
__global__ __launch_bounds__(256) void k_head(const float* __restrict__ pooled,
                                              const float* __restrict__ fcw,
                                              const float* __restrict__ fcb,
                                              float* __restrict__ out) {
  int g = blockIdx.x, tx = threadIdx.x;
  float4 p = {0.f, 0.f, 0.f, 0.f};
  for (int h = tx; h < HH; h += 256) {
    float pv = pooled[(size_t)g * HH + h];
    const float* w = fcw + (size_t)h * CC;
    p.x += pv * w[0]; p.y += pv * w[1]; p.z += pv * w[2]; p.w += pv * w[3];
  }
  __shared__ float4 red[256];
  red[tx] = p; __syncthreads();
  for (int s = 128; s > 0; s >>= 1) {
    if (tx < s) {
      float4 o = red[tx + s];
      red[tx].x += o.x; red[tx].y += o.y; red[tx].z += o.z; red[tx].w += o.w;
    }
    __syncthreads();
  }
  if (tx == 0) {
    float l[CC] = {red[0].x + fcb[0], red[0].y + fcb[1],
                   red[0].z + fcb[2], red[0].w + fcb[3]};
    float m = fmaxf(fmaxf(l[0], l[1]), fmaxf(l[2], l[3]));
    float sum = expf(l[0] - m) + expf(l[1] - m) + expf(l[2] - m) + expf(l[3] - m);
    float lse = m + logf(sum);
    for (int c = 0; c < CC; ++c) out[g * CC + c] = l[c] - lse;
  }
}

// ---------------- host launcher ----------------
extern "C" void kernel_launch(void* const* d_in, const int* in_sizes, int n_in,
                              void* d_out, int out_size, void* d_ws, size_t ws_size,
                              hipStream_t stream) {
  const float* x      = (const float*)d_in[0];
  const float* gcn_w  = (const float*)d_in[1];
  const float* gcn_b  = (const float*)d_in[2];
  const float* sag_w1 = (const float*)d_in[3];
  const float* sag_w2 = (const float*)d_in[4];
  const float* sag_b  = (const float*)d_in[5];
  const float* bn_g   = (const float*)d_in[6];
  const float* bn_b   = (const float*)d_in[7];
  const float* fc_w   = (const float*)d_in[8];
  const float* fc_b   = (const float*)d_in[9];
  const int*   ei     = (const int*)d_in[10];
  float* out = (float*)d_out;

  // workspace carving (256B aligned)
  char* p = (char*)d_ws;
  auto carve = [&](size_t bytes) { void* r = (void*)p; p += (bytes + 255) & ~(size_t)255; return r; };
  unsigned short* xb   = (unsigned short*)carve((size_t)NN * DIN * 2);
  unsigned int*   wp   = (unsigned int*)  carve((size_t)KT * 16 * HH * 4);
  float* hw    = (float*)carve((size_t)NN * HH * 4);
  float* agg   = (float*)carve((size_t)NN * HH * 4);
  float* dinv  = (float*)carve((size_t)NN * 4);
  float* coef  = (float*)carve((size_t)EE * 4);
  float* sc    = (float*)carve((size_t)NN * 4);
  float* t2    = (float*)carve((size_t)NN * 4);
  int*   gidx  = (int*)  carve((size_t)GG * KK * 4);
  float* tanhv = (float*)carve((size_t)GG * KK * 4);
  float* xp    = (float*)carve((size_t)GG * KK * HH * 4);
  float* psum  = (float*)carve((size_t)32 * HH * 4);
  float* psq   = (float*)carve((size_t)32 * HH * 4);
  float* aff   = (float*)carve((size_t)2 * HH * 4);
  float* pooled= (float*)carve((size_t)GG * HH * 4);

  // 0) convert & pack
  k_cvt_x<<<(NN * DIN / 4) / 256, 256, 0, stream>>>(x, xb);
  k_pack_w<<<(KT * 16 * HH) / 256, 256, 0, stream>>>(gcn_w, wp);

  // 1) degree / normalization
  k_deg_init<<<(NN + 255) / 256, 256, 0, stream>>>(dinv);
  k_deg_acc<<<(EE + 255) / 256, 256, 0, stream>>>(ei, dinv);
  k_deg_rsqrt<<<(NN + 255) / 256, 256, 0, stream>>>(dinv);
  k_edge_coef<<<(EE + 255) / 256, 256, 0, stream>>>(ei, dinv, coef);

  // 2) GEMM via WMMA (+ async LDS staging)
  k_gemm_xw<<<dim3(NN / 128, HH / 64), 256, 0, stream>>>(xb, wp, hw);

  // 3) GCN aggregation
  k_selfbias<<<(NN * HH / 4) / 256, 256, 0, stream>>>(hw, dinv, gcn_b, agg);
  k_edge_scatter<<<(EE * (HH / 4)) / 256, 256, 0, stream>>>(ei, coef, hw, agg);

  // 4) ReLU + SAG scores
  k_relu_score<<<NN, 256, 0, stream>>>(agg, sag_w1, sag_w2, sag_b, sc, t2);
  k_score_scatter<<<(EE + 255) / 256, 256, 0, stream>>>(ei, t2, sc);

  // 5) top-k, pooling input
  k_topk<<<GG, 256, 0, stream>>>(sc, gidx, tanhv);
  k_build_xp<<<(GG * KK * (HH / 4)) / 256, 256, 0, stream>>>(agg, gidx, tanhv, xp);

  // 6) BatchNorm stats + affine fold
  k_bn_stage1<<<dim3(32, HH / 256), 256, 0, stream>>>(xp, psum, psq);
  k_bn_stage2<<<HH / 256, 256, 0, stream>>>(psum, psq, bn_g, bn_b, aff);

  // 7) mean pool (+BN affine) and head
  k_pool<<<dim3(GG, HH / 256), 256, 0, stream>>>(xp, aff, pooled);
  k_head<<<GG, 256, 0, stream>>>(pooled, fc_w, fc_b, out);
}
